// imageCaptionModel_61168924230248
// MI455X (gfx1250) — compile-verified, matrix operand order checked by reading, not run on hardware
//
#include <hip/hip_runtime.h>
#include <hip/hip_bf16.h>
#include <math.h>

// ---------------------------------------------------------------------------
// Sizes (fixed by the reference)
// ---------------------------------------------------------------------------
constexpr int kH    = 512;     // hidden
constexpr int kE    = 512;     // embedding
constexpr int kN    = 1024;    // concat width E+H
constexpr int kB    = 128;     // batch
constexpr int kT    = 40;      // time steps
constexpr int kV    = 10000;   // vocab
constexpr int kVpad = 10112;   // vocab padded to 128-col block tiles (79 * 128)
constexpr int kVGG  = 4096;
constexpr int kL    = 2;
constexpr int kWSZ  = kH * kN; // per-layer transposed gate weight elements

typedef __attribute__((ext_vector_type(16))) __bf16       v16bf;
typedef __attribute__((ext_vector_type(8)))  float        v8f;
typedef __attribute__((ext_vector_type(4)))  unsigned int u32x4;

union Frag { v16bf v; u32x4 q[2]; };

// ---------------------------------------------------------------------------
// bf16 fragment loader: per the ISA 16-bit fragment layout (CDNA5 ISA), lane L
// holds row (L%16) with two contiguous 16-byte chunks at K offsets 8*(L/16)
// and 16 + 8*(L/16).  Both A (row-major [MxK]) and pre-transposed B ([NxK])
// use the identical pattern.
// ---------------------------------------------------------------------------
__device__ __forceinline__ Frag load_frag(const __bf16* __restrict__ p) {
  Frag f;
  f.q[0] = *(const u32x4*)(p);
  f.q[1] = *(const u32x4*)(p + 16);
  return f;
}

// One 16x16 output tile with a K-loop (used by the small recurrence GEMMs).
template <int K, int LDA, int LDB>
__device__ __forceinline__ v8f wmma_tile(const __bf16* __restrict__ A,
                                         const __bf16* __restrict__ Bt,
                                         int row0, int col0) {
  const int lane = threadIdx.x & 31;
  const int idx  = lane & 15;
  const int half = lane >> 4;
  const __bf16* ap = A  + (size_t)(row0 + idx) * LDA + half * 8;
  const __bf16* bp = Bt + (size_t)(col0 + idx) * LDB + half * 8;
  v8f acc = {};
#pragma unroll 4
  for (int k = 0; k < K; k += 32) {
    Frag a = load_frag(ap + k);
    Frag b = load_frag(bp + k);
    acc = __builtin_amdgcn_wmma_f32_16x16x32_bf16(false, a.v, false, b.v,
                                                  (short)0, acc, false, false);
  }
  return acc;
}

__device__ __forceinline__ float sigmoidf_(float x) {
  return 1.0f / (1.0f + __expf(-x));
}

// ---------------------------------------------------------------------------
// Setup kernels: convert / transpose / gather into bf16 workspace
// ---------------------------------------------------------------------------
__global__ void k_transpose_bf16(const float* __restrict__ in,  // [K x N] f32
                                 __bf16* __restrict__ out,      // [N x K] bf16
                                 int K, int N) {
  __shared__ float tile[32][33];
  const int kb = blockIdx.y * 32, nb = blockIdx.x * 32;
  for (int i = threadIdx.y; i < 32; i += 8) {
    int k = kb + i, n = nb + threadIdx.x;
    tile[i][threadIdx.x] = (k < K && n < N) ? in[(size_t)k * N + n] : 0.0f;
  }
  __syncthreads();
  for (int i = threadIdx.y; i < 32; i += 8) {
    int n = nb + i, k = kb + threadIdx.x;
    if (n < N && k < K) out[(size_t)n * K + k] = (__bf16)tile[threadIdx.x][i];
  }
}

__global__ void k_cvt_bf16(const float* __restrict__ in, __bf16* __restrict__ out, int n) {
  int i = blockIdx.x * blockDim.x + threadIdx.x;
  if (i < n) out[i] = (__bf16)in[i];
}

// xseq[t][b][e] = bf16(embedding[xTokens[b][t]][e])
__global__ void k_embed(const int* __restrict__ tok, const float* __restrict__ emb,
                        __bf16* __restrict__ xseq) {
  size_t i = (size_t)blockIdx.x * blockDim.x + threadIdx.x;  // over T*B*E
  int e   = (int)(i & (kE - 1));
  size_t row = i >> 9;            // t*B + b
  int b   = (int)(row & (kB - 1));
  int t   = (int)(row >> 7);
  int tk  = tok[b * kT + t];
  xseq[i] = (__bf16)emb[(size_t)tk * kE + e];
}

// ---------------------------------------------------------------------------
// h0 = tanh(vgg @ W_in + b_in), broadcast into both layers' state (f32 + bf16)
// grid: (kH/16), block 256 (8 waves cover all 128 rows)
// ---------------------------------------------------------------------------
__global__ void k_h0(const __bf16* __restrict__ A,   // vgg bf16 [B x VGG]
                     const __bf16* __restrict__ Bt,  // Wt_in    [H x VGG]
                     const float* __restrict__ bias,
                     float* __restrict__ h32, __bf16* __restrict__ hb) {
  const int row0 = (threadIdx.x >> 5) * 16;
  const int col0 = blockIdx.x * 16;
  v8f acc = wmma_tile<kVGG, kVGG, kVGG>(A, Bt, row0, col0);
  const int lane = threadIdx.x & 31, n = lane & 15, half = lane >> 4;
  const int c = col0 + n;
  const float bv = bias[c];
#pragma unroll
  for (int v = 0; v < 8; ++v) {
    int m = row0 + half * 8 + v;
    float val = tanhf(acc[v] + bv);
    h32[m * kH + c]           = val;
    h32[kB * kH + m * kH + c] = val;     // layer 1 copy
    __bf16 hv = (__bf16)val;
    hb[m * kH + c]            = hv;
    hb[kB * kH + m * kH + c]  = hv;
  }
}

// s = [input, h], also seed s1 = [input, _] (second half written by k_gates)
__global__ void k_pack(const __bf16* __restrict__ inp, const __bf16* __restrict__ hbl,
                       __bf16* __restrict__ s, __bf16* __restrict__ s1) {
  int i = blockIdx.x * blockDim.x + threadIdx.x;  // over B*H
  int b = i >> 9, e = i & (kH - 1);
  __bf16 x = inp[i];
  s [b * kN + e]      = x;
  s1[b * kN + e]      = x;
  s [b * kN + kH + e] = hbl[i];
}

// r/u gates: blockIdx.z == 0 -> u (store f32 sigmoid), == 1 -> r (store bf16 r*h
// into second half of s1). grid: (kH/16, 1, 2), block 256.
__global__ void k_gates(const __bf16* __restrict__ s,
                        const __bf16* __restrict__ Wt_u, const __bf16* __restrict__ Wt_r,
                        const float* __restrict__ bu, const float* __restrict__ br,
                        const float* __restrict__ h32l,
                        float* __restrict__ u32, __bf16* __restrict__ s1) {
  const bool is_r = blockIdx.z != 0;            // uniform per block -> EXEC all-1
  const __bf16* Wt  = is_r ? Wt_r : Wt_u;
  const float* bias = is_r ? br : bu;
  const int row0 = (threadIdx.x >> 5) * 16;
  const int col0 = blockIdx.x * 16;
  v8f acc = wmma_tile<kN, kN, kN>(s, Wt, row0, col0);
  const int lane = threadIdx.x & 31, n = lane & 15, half = lane >> 4;
  const int c = col0 + n;
  const float bv = bias[c];
#pragma unroll
  for (int v = 0; v < 8; ++v) {
    int m = row0 + half * 8 + v;
    float g = sigmoidf_(acc[v] + bv);
    if (is_r) s1[m * kN + kH + c] = (__bf16)(g * h32l[m * kH + c]);
    else      u32[m * kH + c]     = g;
  }
}

// c = tanh(s1 @ W_c + b_c); h = u*h + (1-u)*c. Updates f32 + bf16 state, and
// optionally records bf16 h into hseq (layer 1 only) for the big logits GEMM.
__global__ void k_cand(const __bf16* __restrict__ s1, const __bf16* __restrict__ Wt_c,
                       const float* __restrict__ bc, const float* __restrict__ u32,
                       float* __restrict__ h32l, __bf16* __restrict__ hb_l,
                       __bf16* __restrict__ hseq_t) {
  const int row0 = (threadIdx.x >> 5) * 16;
  const int col0 = blockIdx.x * 16;
  v8f acc = wmma_tile<kN, kN, kN>(s1, Wt_c, row0, col0);
  const int lane = threadIdx.x & 31, n = lane & 15, half = lane >> 4;
  const int c = col0 + n;
  const float bv = bc[c];
#pragma unroll
  for (int v = 0; v < 8; ++v) {
    int m = row0 + half * 8 + v;
    float cc = tanhf(acc[v] + bv);
    float u  = u32[m * kH + c];
    float hn = u * h32l[m * kH + c] + (1.0f - u) * cc;
    h32l[m * kH + c] = hn;
    __bf16 hv = (__bf16)hn;
    hb_l[m * kH + c] = hv;
    if (hseq_t) hseq_t[m * kH + c] = hv;
  }
}

// ---------------------------------------------------------------------------
// Big deferred logits GEMM: [T*B, H] @ Wt_out[Vpad x H]^T + b_out -> out[B,T,V]
// Register-blocked: each wave computes a 64x32 output tile (4 row tiles x
// 2 col tiles): per K-step 6 fragment loads feed 8 *independent* WMMAs
// (0.75 loads/WMMA, and 8 interleaved accumulator chains to hide the WMMA
// D->C RAW hazard).  Block = 8 waves (2 row groups x 4 col groups) covering
// 128 rows (= one timestep) x 128 cols; each A row tile is loaded by 4 waves, each B col tile by 2.
// grid: (kVpad/128 = 79, kT = 40), block 256.
// ---------------------------------------------------------------------------
__global__ void k_logits(const __bf16* __restrict__ hseq, const __bf16* __restrict__ Wt_out,
                         const float* __restrict__ bias, float* __restrict__ out) {
  const int wave = threadIdx.x >> 5;
  const int rw   = wave >> 2;                       // 0..1  (row group)
  const int cw   = wave & 3;                        // 0..3  (col group)
  const int lane = threadIdx.x & 31;
  const int idx  = lane & 15;
  const int half = lane >> 4;
  const int rowBase = blockIdx.y * kB + rw * 64;    // row = t*B + b
  const int colBase = blockIdx.x * 128 + cw * 32;

  const __bf16* ap = hseq   + (size_t)(rowBase + idx) * kH + half * 8;
  const __bf16* bp = Wt_out + (size_t)(colBase + idx) * kH + half * 8;

  v8f acc[4][2];
#pragma unroll
  for (int r = 0; r < 4; ++r)
#pragma unroll
    for (int c = 0; c < 2; ++c) acc[r][c] = {};

#pragma unroll 2
  for (int k = 0; k < kH; k += 32) {
    Frag a[4], b[2];
#pragma unroll
    for (int r = 0; r < 4; ++r) a[r] = load_frag(ap + (size_t)r * 16 * kH + k);
#pragma unroll
    for (int c = 0; c < 2; ++c) b[c] = load_frag(bp + (size_t)c * 16 * kH + k);
#pragma unroll
    for (int r = 0; r < 4; ++r)
#pragma unroll
      for (int c = 0; c < 2; ++c)
        acc[r][c] = __builtin_amdgcn_wmma_f32_16x16x32_bf16(
            false, a[r].v, false, b[c].v, (short)0, acc[r][c], false, false);
  }

  const int t = blockIdx.y;
#pragma unroll
  for (int c = 0; c < 2; ++c) {
    const int col = colBase + c * 16 + idx;
    if (col >= kV) continue;                        // padded columns: never stored
    const float bv = bias[col];
#pragma unroll
    for (int r = 0; r < 4; ++r) {
#pragma unroll
      for (int v = 0; v < 8; ++v) {
        int m = rowBase + r * 16 + half * 8 + v;
        int b = m & (kB - 1);
        out[(size_t)b * ((size_t)kT * kV) + (size_t)t * kV + col] = acc[r][c][v] + bv;
      }
    }
  }
}

__global__ void k_hfinal(const float* __restrict__ h32, float* __restrict__ out) {
  int i = blockIdx.x * blockDim.x + threadIdx.x;
  if (i < kL * kB * kH) out[i] = h32[i];
}

// ---------------------------------------------------------------------------
// Host launcher
// ---------------------------------------------------------------------------
extern "C" void kernel_launch(void* const* d_in, const int* in_sizes, int n_in,
                              void* d_out, int out_size, void* d_ws, size_t ws_size,
                              hipStream_t stream) {
  (void)in_sizes; (void)n_in; (void)out_size; (void)ws_size;

  const int*   xTok  = (const int*)  d_in[0];
  const float* vgg   = (const float*)d_in[1];
  const float* emb   = (const float*)d_in[2];
  const float* W_in  = (const float*)d_in[3];
  const float* b_in  = (const float*)d_in[4];
  const float* W_u   = (const float*)d_in[5];
  const float* b_u   = (const float*)d_in[6];
  const float* W_r   = (const float*)d_in[7];
  const float* b_r   = (const float*)d_in[8];
  const float* W_c   = (const float*)d_in[9];
  const float* b_c   = (const float*)d_in[10];
  const float* W_out = (const float*)d_in[11];
  const float* b_out = (const float*)d_in[12];
  float* out = (float*)d_out;

  // --- workspace carve-out (~34 MB) ---
  char* ws = (char*)d_ws;
  size_t off = 0;
  auto carve = [&](size_t bytes) -> void* {
    void* p = ws + off;
    off += (bytes + 255) & ~(size_t)255;
    return p;
  };
  __bf16* Wt_u   = (__bf16*)carve((size_t)kL * kWSZ * 2);         // [l][H][N]
  __bf16* Wt_r   = (__bf16*)carve((size_t)kL * kWSZ * 2);
  __bf16* Wt_c   = (__bf16*)carve((size_t)kL * kWSZ * 2);
  __bf16* Wt_in  = (__bf16*)carve((size_t)kH * kVGG * 2);         // [H][VGG]
  __bf16* Wt_out = (__bf16*)carve((size_t)kVpad * kH * 2);        // [Vpad][H]
  __bf16* vgg_bf = (__bf16*)carve((size_t)kB * kVGG * 2);
  __bf16* xseq   = (__bf16*)carve((size_t)kT * kB * kE * 2);      // [t][b][e]
  __bf16* hseq   = (__bf16*)carve((size_t)kT * kB * kH * 2);      // [t][b][h] (layer 1)
  float*  h32    = (float*) carve((size_t)kL * kB * kH * 4);      // fp32 master state
  __bf16* hb     = (__bf16*)carve((size_t)kL * kB * kH * 2);      // bf16 state
  float*  u32    = (float*) carve((size_t)kB * kH * 4);
  __bf16* sbuf   = (__bf16*)carve((size_t)kB * kN * 2);           // [x, h]
  __bf16* s1buf  = (__bf16*)carve((size_t)kB * kN * 2);           // [x, r*h]

  const dim3 tb(32, 8);

  // --- weight conversion / transposition (L2-resident afterwards) ---
  k_transpose_bf16<<<dim3(kH / 32, kVGG / 32), tb, 0, stream>>>(W_in, Wt_in, kVGG, kH);
  for (int l = 0; l < kL; ++l) {
    k_transpose_bf16<<<dim3(kH / 32, kN / 32), tb, 0, stream>>>(W_u + (size_t)l * kWSZ, Wt_u + (size_t)l * kWSZ, kN, kH);
    k_transpose_bf16<<<dim3(kH / 32, kN / 32), tb, 0, stream>>>(W_r + (size_t)l * kWSZ, Wt_r + (size_t)l * kWSZ, kN, kH);
    k_transpose_bf16<<<dim3(kH / 32, kN / 32), tb, 0, stream>>>(W_c + (size_t)l * kWSZ, Wt_c + (size_t)l * kWSZ, kN, kH);
  }
  k_transpose_bf16<<<dim3((kV + 31) / 32, kH / 32), tb, 0, stream>>>(W_out, Wt_out, kH, kV);
  k_cvt_bf16<<<(kB * kVGG) / 256, 256, 0, stream>>>(vgg, vgg_bf, kB * kVGG);
  k_embed<<<(kT * kB * kE) / 256, 256, 0, stream>>>(xTok, emb, xseq);

  // --- h0 = tanh(vgg @ W_in + b_in) ---
  k_h0<<<kH / 16, 256, 0, stream>>>(vgg_bf, Wt_in, b_in, h32, hb);

  // --- sequential GRU recurrence ---
  const int packBlocks = (kB * kH) / 256;
  for (int t = 0; t < kT; ++t) {
    // layer 0: input = x_t
    k_pack<<<packBlocks, 256, 0, stream>>>(xseq + (size_t)t * kB * kE, hb, sbuf, s1buf);
    k_gates<<<dim3(kH / 16, 1, 2), 256, 0, stream>>>(sbuf, Wt_u, Wt_r, b_u, b_r, h32, u32, s1buf);
    k_cand<<<kH / 16, 256, 0, stream>>>(s1buf, Wt_c, b_c, u32, h32, hb, (__bf16*)nullptr);
    // layer 1: input = h[0]
    k_pack<<<packBlocks, 256, 0, stream>>>(hb, hb + (size_t)kB * kH, sbuf, s1buf);
    k_gates<<<dim3(kH / 16, 1, 2), 256, 0, stream>>>(sbuf, Wt_u + kWSZ, Wt_r + kWSZ,
                                                     b_u + kH, b_r + kH, h32 + (size_t)kB * kH, u32, s1buf);
    k_cand<<<kH / 16, 256, 0, stream>>>(s1buf, Wt_c + kWSZ, b_c + kH, u32,
                                        h32 + (size_t)kB * kH, hb + (size_t)kB * kH,
                                        hseq + (size_t)t * kB * kH);
  }

  // --- one large register-blocked logits GEMM (+bias), permuted to [B,T,V] ---
  k_logits<<<dim3(kVpad / 128, kT), 256, 0, stream>>>(hseq, Wt_out, b_out, out);

  // --- h_final tail of d_out ---
  k_hfinal<<<(kL * kB * kH) / 256, 256, 0, stream>>>(h32, out + (size_t)kB * kT * kV);
}
